// DenoisingBlock_67053029425428
// MI455X (gfx1250) — matrix operand order repaired and checked by reading, not compiled, not run on hardware
//
#include <hip/hip_runtime.h>
#include <cstdint>

// ---------------------------------------------------------------------------
// Non-local attention block for MI455X (gfx1250, wave32, WMMA + TDM).
// All WMMA operands are laid out so each lane's fragment is two contiguous
// 128-bit loads (A: k=16hi+0..15 of row m ; B: k=16hi+0..15 of
// transposed row n).  bf16 inputs, f32 accumulation.
//
// Workspace (requires ~97 MiB):
//   TbT = theta^T * (1/sqrt(D))  bf16 [N][HW][D]   16 MiB
//   PbT = phi^T                  bf16 [N][HW][D]   16 MiB
//   Gb  = x                      bf16 [N][C][HW]   32 MiB
//   GbT = x^T                    bf16 [N][HW][C]   32 MiB
//   Wtb/Wpb/Wfb = weights        bf16              1 MiB
// ---------------------------------------------------------------------------

typedef __attribute__((ext_vector_type(16))) __bf16 v16bf;
typedef __attribute__((ext_vector_type(8)))  __bf16 v8bf;
typedef __attribute__((ext_vector_type(8)))  float  v8f;

#if __has_builtin(__builtin_amdgcn_tensor_load_to_lds) && \
    __has_builtin(__builtin_amdgcn_s_wait_tensorcnt)
#define HAVE_TDM 1
typedef __attribute__((ext_vector_type(4))) unsigned int u32x4;
typedef __attribute__((ext_vector_type(8))) int          i32x8;
typedef __attribute__((ext_vector_type(4))) int          i32x4;
#else
#define HAVE_TDM 0
#endif

#define DEV __device__ __forceinline__

static constexpr int NB  = 32;    // batch
static constexpr int Cc  = 512;   // channels
static constexpr int Dd  = 256;   // embed channels (C/2)
static constexpr int HWl = 1024;  // spatial

DEV unsigned short f2bf_bits(float f) {
  unsigned u = __builtin_bit_cast(unsigned, f);
  unsigned r = u + 0x7FFFu + ((u >> 16) & 1u);   // round-to-nearest-even
  return (unsigned short)(r >> 16);
}
DEV __bf16 f2bf(float f) { return __builtin_bit_cast(__bf16, f2bf_bits(f)); }

DEV v16bf joinbf(v8bf x, v8bf y) {
  return __builtin_shufflevector(x, y, 0, 1, 2, 3, 4, 5, 6, 7,
                                 8, 9, 10, 11, 12, 13, 14, 15);
}
DEV v8f wmma_bf16(v16bf a, v16bf b, v8f c) {
  return __builtin_amdgcn_wmma_f32_16x16x32_bf16(false, a, false, b,
                                                 (short)0, c, false, false);
}

// ---------------------------------------------------------------------------
// f32 -> bf16 pair-packed conversion (weights)
// ---------------------------------------------------------------------------
__global__ void cvt_kernel(const float* __restrict__ in, unsigned* __restrict__ out,
                           long long npairs) {
  long long i = (long long)blockIdx.x * blockDim.x + threadIdx.x;
  long long stride = (long long)gridDim.x * blockDim.x;
  for (; i < npairs; i += stride) {
    unsigned lo = f2bf_bits(in[2 * i]);
    unsigned hi = f2bf_bits(in[2 * i + 1]);
    out[i] = lo | (hi << 16);
  }
}

// ---------------------------------------------------------------------------
// x -> bf16 values, straight (Gb [n][c][q]) and transposed (GbT [n][q][c]),
// via LDS tile so both sides stay coalesced.  block (16,16), grid (HW/16,C/16,N)
// ---------------------------------------------------------------------------
__global__ __launch_bounds__(256)
void xcvt_kernel(const float* __restrict__ x, __bf16* __restrict__ Gb,
                 __bf16* __restrict__ GbT) {
  __shared__ __bf16 tile[16][17];
  const int q0 = blockIdx.x * 16, c0 = blockIdx.y * 16, n = blockIdx.z;
  const int tx = threadIdx.x, ty = threadIdx.y;
  size_t src = ((size_t)n * Cc + c0 + ty) * HWl + q0 + tx;
  __bf16 v = f2bf(x[src]);
  Gb[src] = v;
  tile[ty][tx] = v;
  __syncthreads();
  GbT[((size_t)n * HWl + q0 + ty) * Cc + c0 + tx] = tile[tx][ty];
}

// ---------------------------------------------------------------------------
// Projection GEMM: outT[n][q][d] = bf16( scale * (sum_c W[d][c] x[n][c][q] + b[d]) )
// All operands packed bf16.  One wave per 16x16 tile; grid (HW/16, D/16, N).
// ---------------------------------------------------------------------------
__global__ __launch_bounds__(32)
void proj_kernel(const __bf16* __restrict__ GbT, const __bf16* __restrict__ Wb,
                 const float* __restrict__ bias, __bf16* __restrict__ outT,
                 float scale) {
  const int q0 = blockIdx.x * 16;
  const int d0 = blockIdx.y * 16;
  const int n  = blockIdx.z;
  const int L  = threadIdx.x;
  const int hi = L >> 4, lo = L & 15;

  const __bf16* wr = Wb + (size_t)(d0 + lo) * Cc;                 // A row d
  const __bf16* gt = GbT + ((size_t)n * HWl + q0 + lo) * Cc;      // B row q

  v8f acc = {};
  for (int c0 = 0; c0 < Cc; c0 += 32) {
    v8bf a0 = *(const v8bf*)(wr + c0 + 16 * hi);
    v8bf a1 = *(const v8bf*)(wr + c0 + 16 * hi + 8);
    v8bf b0 = *(const v8bf*)(gt + c0 + 16 * hi);
    v8bf b1 = *(const v8bf*)(gt + c0 + 16 * hi + 8);
    acc = wmma_bf16(joinbf(a0, a1), joinbf(b0, b1), acc);
  }
  // D element r: (m = r + 8*hi, n = lo) -> row q0+lo, cols d0+8hi .. +7
  v8bf o;
#pragma unroll
  for (int r = 0; r < 8; ++r)
    o[r] = f2bf((acc[r] + bias[d0 + 8 * hi + r]) * scale);
  *(v8bf*)(outT + ((size_t)n * HWl + q0 + lo) * Dd + d0 + 8 * hi) = o;
}

// ---------------------------------------------------------------------------
// Flash attention + fuse conv + residual.
// One workgroup (8 waves) per (batch n, 16-query tile q0).
// ---------------------------------------------------------------------------
__global__ __launch_bounds__(256)
void attn_fuse_kernel(const float* __restrict__ x,
                      const __bf16* __restrict__ TbT,
                      const __bf16* __restrict__ PbT,
                      const __bf16* __restrict__ Gb,
                      const __bf16* __restrict__ Wfb,
                      const float* __restrict__ bfuse,
                      float* __restrict__ outp) {
  __shared__ alignas(16) __bf16 sT[16][Dd];      // theta^T tile [q][d]   8 KiB
  __shared__ alignas(16) float  sS[16][128];     // score chunk [q][k]    8 KiB
  __shared__ alignas(16) __bf16 sPq[16][128];    // probs [q][k]          4 KiB
  __shared__ alignas(16) __bf16 sAttnT[16][Cc];  // attn^T [q][c]        16 KiB
  __shared__ float sM[16], sL[16], sFac[16];

  const int n    = blockIdx.y;
  const int q0   = blockIdx.x * 16;
  const int tid  = threadIdx.x;
  const int wave = tid >> 5;
  const int L    = tid & 31;
  const int hi   = L >> 4, lo = L & 15;

  // ---- Stage theta tile (16x256 bf16, contiguous 8 KiB block) into LDS.
#if HAVE_TDM
  // Tensor Data Mover: wave 0 issues one async DMA descriptor (D#) for the
  // whole tile; the workgroup barrier below orders it for the other waves.
  if (wave == 0) {
    const __bf16* gsrc = TbT + ((size_t)n * HWl + q0) * Dd;
    unsigned long long ga = (unsigned long long)(uintptr_t)gsrc;
    unsigned lds = (unsigned)(uintptr_t)&sT[0][0];  // flat[31:0] == LDS offset
    u32x4 g0;
    g0[0] = 1u;                                       // count=1, user D#
    g0[1] = lds;                                      // lds_addr
    g0[2] = (unsigned)ga;                             // global_addr[31:0]
    g0[3] = (unsigned)((ga >> 32) & 0x01FFFFFFu) |    // global_addr[56:32]
            (2u << 30);                               // type=2 ("image")
    i32x8 g1;
    g1[0] = 0x00010000;          // workgroup_mask=0, data_size=1 (2 bytes)
    g1[1] = (int)(4096u << 16);  // tensor_dim0 = 4096 elements (bits 63:48)
    g1[2] = 0x00010000;          // tensor_dim0 hi=0, tensor_dim1 = 1
    g1[3] = (int)(4096u << 16);  // tensor_dim1 hi=0, tile_dim0 = 4096
    g1[4] = 0x00000001;          // tile_dim1 = 1, tile_dim2 = 0
    g1[5] = 4096;                // tensor_dim0_stride lo
    g1[6] = 0;
    g1[7] = 0;
    i32x4 z4 = {};
    i32x8 z8 = {};
    __builtin_amdgcn_tensor_load_to_lds(g0, g1, z4, z4, z8, 0);
    __builtin_amdgcn_s_wait_tensorcnt(0);
  }
#else
  {
    const v8bf* src = (const v8bf*)(TbT + ((size_t)n * HWl + q0) * Dd);
    v8bf* dst = (v8bf*)&sT[0][0];
    for (int i = tid; i < (16 * Dd) / 8; i += 256) dst[i] = src[i];
  }
#endif
  if (tid < 16) { sM[tid] = -__builtin_inff(); sL[tid] = 0.0f; }
  __syncthreads();

  v8f accO[4];  // this wave's 64 channels: c = wave*64 + t*16 + (r+8*hi)
#pragma unroll
  for (int t = 0; t < 4; ++t) { v8f z = {}; accO[t] = z; }

  const __bf16* Pn = PbT + (size_t)n * HWl * Dd;
  const __bf16* Gn = Gb + (size_t)n * Cc * HWl;

  for (int kc = 0; kc < HWl; kc += 128) {
    // Prefetch next chunk's phi row and value row for this wave.
    if (kc + 128 < HWl) {
      __builtin_prefetch(Pn + (size_t)(kc + 128 + wave * 16 + lo) * Dd, 0, 0);
      __builtin_prefetch(Gn + (size_t)(wave * 64 + lo) * HWl + kc + 128, 0, 0);
    }

    // --- 1. scores: wave w owns k-tile kc + 16w.  S = theta^T phi (K=256).
    {
      const int k0 = kc + wave * 16;
      const __bf16* pr = Pn + (size_t)(k0 + lo) * Dd;  // B row k
      v8f s = {};
      for (int dd = 0; dd < Dd; dd += 32) {
        v8bf a0 = *(const v8bf*)&sT[lo][dd + 16 * hi];
        v8bf a1 = *(const v8bf*)&sT[lo][dd + 16 * hi + 8];
        v8bf b0 = *(const v8bf*)(pr + dd + 16 * hi);
        v8bf b1 = *(const v8bf*)(pr + dd + 16 * hi + 8);
        s = wmma_bf16(joinbf(a0, a1), joinbf(b0, b1), s);
      }
#pragma unroll
      for (int r = 0; r < 8; ++r)
        sS[r + 8 * hi][wave * 16 + lo] = s[r];   // scale folded into theta
    }
    __syncthreads();

    // --- 2. online-softmax running max + rescale factor
    if (tid < 16) {
      float mold = sM[tid], cmax = -__builtin_inff();
      for (int j = 0; j < 128; ++j) cmax = fmaxf(cmax, sS[tid][j]);
      float mnew = fmaxf(mold, cmax);
      sFac[tid] = __expf(mold - mnew);
      sM[tid]   = mnew;
    }
    __syncthreads();

    // --- 3. exponentiate: f32 back to sS (row sums), packed bf16 to sPq
    {
      int row = tid >> 4, cb = (tid & 15) * 8;
      float mnew = sM[row];
      v8bf p8;
#pragma unroll
      for (int j = 0; j < 8; ++j) {
        float p = __expf(sS[row][cb + j] - mnew);
        sS[row][cb + j] = p;
        p8[j] = f2bf(p);
      }
      *(v8bf*)&sPq[row][cb] = p8;
    }
    __syncthreads();

    // --- 4. deterministic row-sum update (16 threads, overlapped with 5)
    if (tid < 16) {
      float sum = 0.0f;
      for (int j = 0; j < 128; ++j) sum += sS[tid][j];
      sL[tid] = sL[tid] * sFac[tid] + sum;
    }

    // --- 5. rescale accumulators, add chunk: O[c,q] += g[c,k] p~[k,q]
    {
      float fac = sFac[lo];
#pragma unroll
      for (int t = 0; t < 4; ++t) {
        const int c0 = wave * 64 + t * 16;
        const __bf16* gr = Gn + (size_t)(c0 + lo) * HWl + kc;  // A row c
#pragma unroll
        for (int r = 0; r < 8; ++r) accO[t][r] *= fac;
#pragma unroll
        for (int kk = 0; kk < 128; kk += 32) {
          v8bf a0 = *(const v8bf*)(gr + kk + 16 * hi);
          v8bf a1 = *(const v8bf*)(gr + kk + 16 * hi + 8);
          v8bf b0 = *(const v8bf*)&sPq[lo][kk + 16 * hi];
          v8bf b1 = *(const v8bf*)&sPq[lo][kk + 16 * hi + 8];
          accO[t] = wmma_bf16(joinbf(a0, a1), joinbf(b0, b1), accO[t]);
        }
      }
    }
    __syncthreads();
  }

  // --- finalize: attn = accO / l -> LDS bf16 [q][c], packed 16B stores
  {
    float linv = 1.0f / sL[lo];
#pragma unroll
    for (int t = 0; t < 4; ++t) {
      v8bf o;
#pragma unroll
      for (int r = 0; r < 8; ++r) o[r] = f2bf(accO[t][r] * linv);
      *(v8bf*)&sAttnT[lo][wave * 64 + t * 16 + 8 * hi] = o;
    }
  }
  __syncthreads();

  // --- fuse 1x1 conv + residual: out = x + W_fuse @ attn + b_fuse (K=512)
#pragma unroll
  for (int t = 0; t < 4; ++t) {
    const int o0 = wave * 64 + t * 16;
    const __bf16* wr = Wfb + (size_t)(o0 + lo) * Cc;  // A row o
    v8f acc = {};
    for (int c0 = 0; c0 < Cc; c0 += 32) {
      v8bf a0 = *(const v8bf*)(wr + c0 + 16 * hi);
      v8bf a1 = *(const v8bf*)(wr + c0 + 16 * hi + 8);
      v8bf b0 = *(const v8bf*)&sAttnT[lo][c0 + 16 * hi];
      v8bf b1 = *(const v8bf*)&sAttnT[lo][c0 + 16 * hi + 8];
      acc = wmma_bf16(joinbf(a0, a1), joinbf(b0, b1), acc);
    }
#pragma unroll
    for (int r = 0; r < 8; ++r) {
      int o = o0 + r + 8 * hi;
      size_t idx = ((size_t)n * Cc + o) * HWl + q0 + lo;
      outp[idx] = x[idx] + bfuse[o] + acc[r];
    }
  }
}

// ---------------------------------------------------------------------------
extern "C" void kernel_launch(void* const* d_in, const int* in_sizes, int n_in,
                              void* d_out, int out_size, void* d_ws, size_t ws_size,
                              hipStream_t stream) {
  (void)in_sizes; (void)n_in; (void)out_size; (void)ws_size;
  const float* x   = (const float*)d_in[0];
  const float* Wt  = (const float*)d_in[1];
  const float* bt  = (const float*)d_in[2];
  const float* Wp  = (const float*)d_in[3];
  const float* bp  = (const float*)d_in[4];
  const float* Wf  = (const float*)d_in[5];
  const float* bf_ = (const float*)d_in[6];
  float* outp = (float*)d_out;

  __bf16* TbT = (__bf16*)d_ws;                          // [N][HW][D]  16 MiB
  __bf16* PbT = TbT + (size_t)NB * HWl * Dd;            // [N][HW][D]  16 MiB
  __bf16* Gb  = PbT + (size_t)NB * HWl * Dd;            // [N][C][HW]  32 MiB
  __bf16* GbT = Gb + (size_t)NB * Cc * HWl;             // [N][HW][C]  32 MiB
  __bf16* Wtb = GbT + (size_t)NB * HWl * Cc;            // [D][C]
  __bf16* Wpb = Wtb + (size_t)Dd * Cc;                  // [D][C]
  __bf16* Wfb = Wpb + (size_t)Dd * Cc;                  // [C][C]

  // Weight conversions (packed u32 stores).
  cvt_kernel<<<128, 256, 0, stream>>>(Wt, (unsigned*)Wtb, (long long)Dd * Cc / 2);
  cvt_kernel<<<128, 256, 0, stream>>>(Wp, (unsigned*)Wpb, (long long)Dd * Cc / 2);
  cvt_kernel<<<256, 256, 0, stream>>>(Wf, (unsigned*)Wfb, (long long)Cc * Cc / 2);

  // Values: straight + transposed bf16 copies of x.
  dim3 gx(HWl / 16, Cc / 16, NB);
  xcvt_kernel<<<gx, dim3(16, 16), 0, stream>>>(x, Gb, GbT);

  // Projections (scale 1/sqrt(D) folded into theta).
  dim3 gp(HWl / 16, Dd / 16, NB);
  proj_kernel<<<gp, 32, 0, stream>>>(GbT, Wtb, bt, TbT, 0.0625f);
  proj_kernel<<<gp, 32, 0, stream>>>(GbT, Wpb, bp, PbT, 1.0f);

  // Attention + fuse.
  dim3 ga(HWl / 16, NB);
  attn_fuse_kernel<<<ga, 256, 0, stream>>>(x, TbT, PbT, Gb, Wfb, bf_, outp);
}
